// MultiheadAttention_11570641895991
// MI455X (gfx1250) — matrix operand order repaired and checked by reading, not compiled, not run on hardware
//
#include <hip/hip_runtime.h>
#include <stdint.h>

// ---------------- Types ----------------
typedef __attribute__((ext_vector_type(16))) __bf16 v16bf;
typedef __attribute__((ext_vector_type(8)))  float  v8f;
typedef unsigned short u16;

union BF16x16 { v16bf v; uint4 q[2]; };

__device__ __forceinline__ u16 f2bf(float f) {
    union { __bf16 b; u16 u; } c; c.b = (__bf16)f; return c.u;
}

__device__ __forceinline__ v8f wmma_bf16(const BF16x16& a, const BF16x16& b, v8f c) {
    return __builtin_amdgcn_wmma_f32_16x16x32_bf16(false, a.v, false, b.v,
                                                   (short)0, c, false, false);
}

// ---------------- Constants ----------------
#define BB   4
#define SS   2048
#define DD   768
#define HH   16
#define DHH  48
#define NEGV (-1.0e9f)
#define EPSV (1e-3f)
#define SCL  (0.14433756729740643f) /* 1/sqrt(48) */

// =====================================================================
// Kernel 0a: elementwise f32 -> bf16 (activations), float4 -> 4x bf16
// =====================================================================
__global__ __launch_bounds__(256) void cvt_bf16(const float* __restrict__ X,
                                                u16* __restrict__ Y)
{
    size_t i = ((size_t)blockIdx.x * 256 + threadIdx.x) * 4;
    float4 f = *(const float4*)(X + i);
    union { u16 u[4]; uint2 v2; } pk;
    pk.u[0] = f2bf(f.x); pk.u[1] = f2bf(f.y);
    pk.u[2] = f2bf(f.z); pk.u[3] = f2bf(f.w);
    *(uint2*)(Y + i) = pk.v2;
}

// =====================================================================
// Kernel 0b: W[768,768] f32 -> Wt[n][k] bf16 (transposed), LDS-tiled
// =====================================================================
__global__ __launch_bounds__(256) void cvt_wT(const float* __restrict__ W,
                                              u16* __restrict__ Wt)
{
    __shared__ float t[32][33];
    const int tx = threadIdx.x & 31;       // 0..31
    const int ty = threadIdx.x >> 5;       // 0..7
    const int n0 = blockIdx.x * 32;        // source col block
    const int k0 = blockIdx.y * 32;        // source row block
#pragma unroll
    for (int i = 0; i < 4; ++i) {
        int k = k0 + ty + i * 8;
        t[ty + i * 8][tx] = W[(size_t)k * DD + n0 + tx];
    }
    __syncthreads();
#pragma unroll
    for (int i = 0; i < 4; ++i) {
        int n = n0 + ty + i * 8;
        Wt[(size_t)n * DD + k0 + tx] = f2bf(t[tx][ty + i * 8]);
    }
}

// =====================================================================
// Kernel 1: projection GEMM  Out[M,768](bf16) = A @ W + b, bf16 WMMA.
// Operands pre-converted (A bf16 row-major, W bf16 pre-transposed), so
// fragments load straight from global/L2: no LDS, no barriers.
// 128x128 tile per 256-thread block; 8 waves x (32x64); K-step 32.
// =====================================================================
__global__ __launch_bounds__(256) void proj_gemm(const u16* __restrict__ Abf,
                                                 const u16* __restrict__ Wt,
                                                 const float* __restrict__ bias,
                                                 u16* __restrict__ Out)
{
    const int N = DD, K = DD;
    const int tid  = threadIdx.x;
    const int lane = tid & 31;
    const int w    = tid >> 5;
    const int half = (lane >> 4) & 1;
    const int l16  = lane & 15;
    const int wm   = w & 3;   // 0..3 -> 32-row quadrant
    const int wn   = w >> 2;  // 0..1 -> 64-col half
    const int mbase = blockIdx.y * 128;
    const int nbase = blockIdx.x * 128;

    // per-lane base pointers (A rows / Wt cols), fragment-layout aligned
    const u16* ar[2];
    const u16* wc[4];
#pragma unroll
    for (int mb = 0; mb < 2; ++mb)
        ar[mb] = Abf + (size_t)(mbase + wm * 32 + mb * 16 + l16) * K;
#pragma unroll
    for (int nb = 0; nb < 4; ++nb)
        wc[nb] = Wt + (size_t)(nbase + wn * 64 + nb * 16 + l16) * K;

    v8f c[2][4];
#pragma unroll
    for (int mb = 0; mb < 2; ++mb)
#pragma unroll
        for (int nb = 0; nb < 4; ++nb) c[mb][nb] = v8f{};

    for (int k0 = 0; k0 < K; k0 += 32) {
        if (k0 + 32 < K) { // pull next k-slab toward L0/L2 (global_prefetch_b8)
#pragma unroll
            for (int mb = 0; mb < 2; ++mb)
                __builtin_prefetch(ar[mb] + k0 + 32, 0, 3);
#pragma unroll
            for (int nb = 0; nb < 4; ++nb)
                __builtin_prefetch(wc[nb] + k0 + 32, 0, 3);
        }
        BF16x16 af[2], bf[4];
#pragma unroll
        for (int mb = 0; mb < 2; ++mb) {
            af[mb].q[0] = *(const uint4*)(ar[mb] + k0 + half * 8);
            af[mb].q[1] = *(const uint4*)(ar[mb] + k0 + 16 + half * 8);
        }
#pragma unroll
        for (int nb = 0; nb < 4; ++nb) {
            bf[nb].q[0] = *(const uint4*)(wc[nb] + k0 + half * 8);
            bf[nb].q[1] = *(const uint4*)(wc[nb] + k0 + 16 + half * 8);
        }
#pragma unroll
        for (int mb = 0; mb < 2; ++mb)
#pragma unroll
            for (int nb = 0; nb < 4; ++nb)
                c[mb][nb] = wmma_bf16(af[mb], bf[nb], c[mb][nb]);
    }

    // ---- epilogue: bias + bf16 store ----
#pragma unroll
    for (int mb = 0; mb < 2; ++mb)
#pragma unroll
        for (int nb = 0; nb < 4; ++nb) {
            int col = nbase + wn * 64 + nb * 16 + l16;
            float bv = bias[col];
#pragma unroll
            for (int r = 0; r < 8; ++r) {
                int row = mbase + wm * 32 + mb * 16 + r + half * 8;
                Out[(size_t)row * N + col] = f2bf(c[mb][nb][r] + bv);
            }
        }
}

// =====================================================================
// Kernel 2: causal flash attention. One wave per (b, h, 32 query rows).
// DH=48 zero-padded to 64 for QK^T (two bf16 WMMAs); PV has K=32 exact.
// =====================================================================
__global__ __launch_bounds__(32) void attn_kernel(const u16* __restrict__ Qb,
                                                  const u16* __restrict__ Kb,
                                                  const u16* __restrict__ Vb,
                                                  float* __restrict__ AO)
{
    __shared__ __align__(16) u16 sP[32 * 32]; // P tile bf16, stride 32
    __shared__ __align__(16) u16 sV[48 * 32]; // V^T tile bf16: sV[d][j], stride 32

    const int lane = threadIdx.x;
    const int half = lane >> 4;
    const int l16  = lane & 15;
    const int qt   = blockIdx.x & 63;   // S/32 = 64 query tiles
    const int bh   = blockIdx.x >> 6;
    const int b    = bh >> 4;
    const int h    = bh & 15;
    const int qbase = qt * 32;

    const uint4 zq = {0u, 0u, 0u, 0u};

    // ---- Q A-fragments (held for the whole kernel) ----
    BF16x16 qf[2][2];
#pragma unroll
    for (int mb = 0; mb < 2; ++mb) {
        int row = qbase + mb * 16 + l16;
        const u16* qr = Qb + ((size_t)b * SS + row) * DD + h * DHH;
        qf[mb][0].q[0] = *(const uint4*)(qr + half * 8);        // d 0..7 / 8..15
        qf[mb][0].q[1] = *(const uint4*)(qr + 16 + half * 8);   // d 16..23 / 24..31
        qf[mb][1].q[0] = *(const uint4*)(qr + 32 + half * 8);   // d 32..39 / 40..47
        qf[mb][1].q[1] = zq;                                    // d 48..63 pad
    }

    v8f o[2][3];
    float mst[2][8], lst[2][8];
#pragma unroll
    for (int mb = 0; mb < 2; ++mb) {
#pragma unroll
        for (int nt = 0; nt < 3; ++nt) o[mb][nt] = v8f{};
#pragma unroll
        for (int r = 0; r < 8; ++r) { mst[mb][r] = -3.0e38f; lst[mb][r] = 0.0f; }
    }

    for (int kt = 0; kt <= qt; ++kt) {
        const int kbase = kt * 32;

        // ---- stage V tile transposed into LDS ----
        {
            const int j = lane;
            const uint4* vp = (const uint4*)(Vb + ((size_t)b * SS + kbase + j) * DD + h * DHH);
            union { uint4 q[6]; u16 u[48]; } vv;
#pragma unroll
            for (int i = 0; i < 6; ++i) vv.q[i] = vp[i];
#pragma unroll
            for (int d = 0; d < 48; ++d) sV[d * 32 + j] = vv.u[d];
        }

        // ---- K B-fragments (d-contiguous from global) ----
        BF16x16 kf[2][2];
#pragma unroll
        for (int nb = 0; nb < 2; ++nb) {
            int j = kbase + nb * 16 + l16;
            const u16* kr = Kb + ((size_t)b * SS + j) * DD + h * DHH;
            kf[nb][0].q[0] = *(const uint4*)(kr + half * 8);        // d 0..7 / 8..15
            kf[nb][0].q[1] = *(const uint4*)(kr + 16 + half * 8);   // d 16..23 / 24..31
            kf[nb][1].q[0] = *(const uint4*)(kr + 32 + half * 8);   // d 32..39 / 40..47
            kf[nb][1].q[1] = zq;                                    // d 48..63 pad
        }

        // ---- scores S = Q K^T (padded K=64) ----
        v8f s[2][2];
#pragma unroll
        for (int mb = 0; mb < 2; ++mb)
#pragma unroll
            for (int nb = 0; nb < 2; ++nb) {
                v8f acc = v8f{};
                acc = wmma_bf16(qf[mb][0], kf[nb][0], acc);
                acc = wmma_bf16(qf[mb][1], kf[nb][1], acc);
                s[mb][nb] = acc;
            }

        // ---- scale + causal mask (diagonal tile only) ----
        const bool diag = (kt == qt);
#pragma unroll
        for (int mb = 0; mb < 2; ++mb)
#pragma unroll
            for (int nb = 0; nb < 2; ++nb)
#pragma unroll
                for (int r = 0; r < 8; ++r) {
                    float val = s[mb][nb][r] * SCL;
                    if (diag) {
                        int mrel = mb * 16 + r + half * 8;
                        int jrel = nb * 16 + l16;
                        if (jrel > mrel) val = NEGV;
                    }
                    s[mb][nb][r] = val;
                }

        // ---- online softmax (row = 16 lanes of one VGPR) ----
#pragma unroll
        for (int mb = 0; mb < 2; ++mb) {
#pragma unroll
            for (int r = 0; r < 8; ++r) {
                float rm = fmaxf(s[mb][0][r], s[mb][1][r]);
#pragma unroll
                for (int off = 8; off >= 1; off >>= 1)
                    rm = fmaxf(rm, __shfl_xor(rm, off, 32));
                float mnew = fmaxf(mst[mb][r], rm);
                float corr = __expf(mst[mb][r] - mnew);
                float p0 = __expf(s[mb][0][r] - mnew);
                float p1 = __expf(s[mb][1][r] - mnew);
                float rs = p0 + p1;
#pragma unroll
                for (int off = 8; off >= 1; off >>= 1)
                    rs += __shfl_xor(rs, off, 32);
                lst[mb][r] = lst[mb][r] * corr + rs;
                mst[mb][r] = mnew;
#pragma unroll
                for (int nt = 0; nt < 3; ++nt) o[mb][nt][r] *= corr;
                s[mb][0][r] = p0;
                s[mb][1][r] = p1;
            }
        }

        // ---- P (C-layout) -> LDS bf16 ----
#pragma unroll
        for (int mb = 0; mb < 2; ++mb)
#pragma unroll
            for (int nb = 0; nb < 2; ++nb)
#pragma unroll
                for (int r = 0; r < 8; ++r)
                    sP[(mb * 16 + r + half * 8) * 32 + nb * 16 + l16] = f2bf(s[mb][nb][r]);
        __syncthreads(); // single-wave workgroup: barrier ~ S_NOP + DS fence

        // ---- P A-fragments + V B-fragments from LDS ----
        BF16x16 pf[2], vf[3];
#pragma unroll
        for (int mb = 0; mb < 2; ++mb) {
            int row = mb * 16 + l16;
            pf[mb].q[0] = *(const uint4*)&sP[row * 32 + half * 8];
            pf[mb].q[1] = *(const uint4*)&sP[row * 32 + 16 + half * 8];
        }
#pragma unroll
        for (int nt = 0; nt < 3; ++nt) {
            int d = nt * 16 + l16;
            vf[nt].q[0] = *(const uint4*)&sV[d * 32 + half * 8];
            vf[nt].q[1] = *(const uint4*)&sV[d * 32 + 16 + half * 8];
        }
#pragma unroll
        for (int mb = 0; mb < 2; ++mb)
#pragma unroll
            for (int nt = 0; nt < 3; ++nt)
                o[mb][nt] = wmma_bf16(pf[mb], vf[nt], o[mb][nt]);
        __syncthreads();
    }

    // ---- normalize + store f32 ----
#pragma unroll
    for (int mb = 0; mb < 2; ++mb)
#pragma unroll
        for (int nt = 0; nt < 3; ++nt)
#pragma unroll
            for (int r = 0; r < 8; ++r) {
                int row = qbase + mb * 16 + r + half * 8;
                int col = h * DHH + nt * 16 + l16;
                AO[((size_t)b * SS + row) * DD + col] = o[mb][nt][r] / lst[mb][r];
            }
}

// =====================================================================
// Kernel 3: LayerNorm over D=768, one 256-thread block per row.
// =====================================================================
__global__ __launch_bounds__(256) void ln_kernel(const float* __restrict__ X,
                                                 const float* __restrict__ gamma,
                                                 const float* __restrict__ beta,
                                                 float* __restrict__ out)
{
    __shared__ float r1[256], r2[256];
    const int row = blockIdx.x;
    const int tid = threadIdx.x;
    const float* x = X + (size_t)row * DD;
    float v0 = x[tid], v1 = x[tid + 256], v2 = x[tid + 512];
    r1[tid] = v0 + v1 + v2;
    r2[tid] = v0 * v0 + v1 * v1 + v2 * v2;
    __syncthreads();
    for (int off = 128; off > 0; off >>= 1) {
        if (tid < off) { r1[tid] += r1[tid + off]; r2[tid] += r2[tid + off]; }
        __syncthreads();
    }
    const float inv = 1.0f / (float)DD;
    float mu  = r1[0] * inv;
    float var = r2[0] * inv - mu * mu;
    float rs  = rsqrtf(var + EPSV);
    float* y = out + (size_t)row * DD;
    y[tid]       = (v0 - mu) * rs * gamma[tid]       + beta[tid];
    y[tid + 256] = (v1 - mu) * rs * gamma[tid + 256] + beta[tid + 256];
    y[tid + 512] = (v2 - mu) * rs * gamma[tid + 512] + beta[tid + 512];
}

// =====================================================================
// Launch
// =====================================================================
extern "C" void kernel_launch(void* const* d_in, const int* in_sizes, int n_in,
                              void* d_out, int out_size, void* d_ws, size_t ws_size,
                              hipStream_t stream)
{
    (void)in_sizes; (void)n_in; (void)out_size; (void)ws_size;
    const float* q     = (const float*)d_in[0];
    const float* k     = (const float*)d_in[1];
    const float* v     = (const float*)d_in[2];
    const float* Wq    = (const float*)d_in[3];
    const float* bq    = (const float*)d_in[4];
    const float* Wk    = (const float*)d_in[5];
    const float* bk    = (const float*)d_in[6];
    const float* Wv    = (const float*)d_in[7];
    const float* bv    = (const float*)d_in[8];
    const float* gamma = (const float*)d_in[9];
    const float* beta  = (const float*)d_in[10];

    const size_t tot  = (size_t)BB * SS * DD; // 8192*768 activations
    const size_t wtot = (size_t)DD * DD;      // 768*768 weights

    u16* qbf = (u16*)d_ws;          // converted inputs
    u16* kbf = qbf + tot;
    u16* vbf = kbf + tot;
    u16* WqT = vbf + tot;           // converted + transposed weights
    u16* WkT = WqT + wtot;
    u16* WvT = WkT + wtot;
    u16* Qbf = WvT + wtot;          // projected Q/K/V (bf16)
    u16* Kbf = Qbf + tot;
    u16* Vbf = Kbf + tot;
    float* AO = (float*)(Vbf + tot);

    // 0) one-time precision / layout conversion
    const int cblk = (int)(tot / 4 / 256); // 6144
    cvt_bf16<<<cblk, 256, 0, stream>>>(q, qbf);
    cvt_bf16<<<cblk, 256, 0, stream>>>(k, kbf);
    cvt_bf16<<<cblk, 256, 0, stream>>>(v, vbf);
    dim3 tgrid(DD / 32, DD / 32); // 24 x 24
    cvt_wT<<<tgrid, 256, 0, stream>>>(Wq, WqT);
    cvt_wT<<<tgrid, 256, 0, stream>>>(Wk, WkT);
    cvt_wT<<<tgrid, 256, 0, stream>>>(Wv, WvT);

    // 1) projections (barrier-free WMMA GEMM)
    dim3 pgrid(DD / 128, (BB * SS) / 128); // 6 x 64
    proj_gemm<<<pgrid, 256, 0, stream>>>(qbf, WqT, bq, Qbf);
    proj_gemm<<<pgrid, 256, 0, stream>>>(kbf, WkT, bk, Kbf);
    proj_gemm<<<pgrid, 256, 0, stream>>>(vbf, WvT, bv, Vbf);

    // 2) causal flash attention
    attn_kernel<<<BB * HH * (SS / 32), 32, 0, stream>>>(Qbf, Kbf, Vbf, AO);

    // 3) layer norm
    ln_kernel<<<BB * SS, 256, 0, stream>>>(AO, gamma, beta, (float*)d_out);
}
